// AttentionPruneViTSmall_32856499814533
// MI455X (gfx1250) — matrix operand order repaired and checked by reading, not compiled, not run on hardware
//
#include <hip/hip_runtime.h>
#include <hip/hip_bf16.h>
#include <math.h>

typedef __bf16 bf16_t;
typedef __attribute__((ext_vector_type(16))) __bf16 v16bf;
typedef __attribute__((ext_vector_type(8)))  __bf16 v8bf;
typedef __attribute__((ext_vector_type(8)))  float   v8f;

#define NB      64
#define CDIM    384
#define NH      6
#define HDIM    64
#define MLPD    1536
#define NLAYER  12
#define NP0     197
#define NPATCH_ 196
#define PDIM    768
#define NCLS_   100

// ---------------------------------------------------------------------------
// Batched WMMA GEMM with register blocking: each wave computes a 16 x (16*NT)
// strip of D.  A fragment loaded once per K-step, reused by NT independent
// accumulator chains (hides WMMA RAW latency, cuts A traffic NT x).
// TRANSB=1: B stored [Ncols x K] row-major (Bp[n*ldb + k])  -> contiguous K
// TRANSB=0: B stored [K x Ncols] row-major (Bp[k*ldb + n])
// flags: 1=bias(fp32, per-col)  2=GELU(exact erf)  4=residual add (fp32)
// per-z offsets: off = (z/zdiv)*s1 + (z%zdiv)*s2 for A, B, D independently.
// ---------------------------------------------------------------------------
template<int TRANSB, int NT>
__global__ __launch_bounds__(32)
void wmma_gemm_bf16(const bf16_t* A, const bf16_t* Bp, const float* bias,
                    const float* resid, float* Df, bf16_t* Db,
                    int M, int Ncols, int K, int lda, int ldb, int ldd,
                    int flags, int zdiv,
                    long sA1, long sA2, long sB1, long sB2, long sD1, long sD2)
{
    const int lane = threadIdx.x;
    const int half = lane >> 4;
    const int l16  = lane & 15;
    const int n0 = blockIdx.x * (16 * NT);
    const int m0 = blockIdx.y << 4;
    const int z  = blockIdx.z;
    const int z1 = z / zdiv;
    const int z2 = z - z1 * zdiv;
    A  += z1 * sA1 + z2 * sA2;
    Bp += z1 * sB1 + z2 * sB2;
    const long dOff = z1 * sD1 + z2 * sD2;

    int rowA = m0 + l16; if (rowA > M - 1) rowA = M - 1;   // clamp (stores guarded)
    const bf16_t* Arow = A + (long)rowA * lda;

    int  colB[NT];                 // clamped column per sub-tile
    bool colOk[NT];
    const bf16_t* Bcol[NT];        // used when TRANSB==1
    #pragma unroll
    for (int t = 0; t < NT; ++t) {
        int c = n0 + t * 16 + l16;
        colOk[t] = c < Ncols;
        if (c > Ncols - 1) c = Ncols - 1;
        colB[t] = c;
        Bcol[t] = Bp + (long)c * ldb;
    }

    v8f acc[NT];
    #pragma unroll
    for (int t = 0; t < NT; ++t)
        acc[t] = (v8f){0.f, 0.f, 0.f, 0.f, 0.f, 0.f, 0.f, 0.f};

    const int K32 = K & ~31;
    union V16 { v16bf v; v8bf h[2]; };

    for (int k0 = 0; k0 < K32; k0 += 32) {
        V16 a;
        // A operand, contiguous k order shared with B: lane-half h holds K = k0+16h+{0..15}
        a.h[0] = *(const v8bf*)(Arow + k0 + 16 * half);
        a.h[1] = *(const v8bf*)(Arow + k0 + 16 * half + 8);
        V16 b[NT];
        #pragma unroll
        for (int t = 0; t < NT; ++t) {
            if (TRANSB) {
                // B operand, contiguous k order shared with A: lane-half h holds K = k0+16h+{0..15}
                b[t].h[0] = *(const v8bf*)(Bcol[t] + k0 + 16 * half);
                b[t].h[1] = *(const v8bf*)(Bcol[t] + k0 + 16 * half + 8);
            } else {
                #pragma unroll
                for (int j = 0; j < 16; ++j)
                    b[t].v[j] = Bp[(long)(k0 + 16 * half + j) * ldb + colB[t]];
            }
        }
        if (k0 + 96 < K32) {
            __builtin_prefetch(Arow + k0 + 96, 0, 3);
            if (TRANSB) {
                #pragma unroll
                for (int t = 0; t < NT; ++t)
                    __builtin_prefetch(Bcol[t] + k0 + 96, 0, 3);
            }
        }
        #pragma unroll
        for (int t = 0; t < NT; ++t)     // NT independent WMMA chains
            acc[t] = __builtin_amdgcn_wmma_f32_16x16x32_bf16(false, a.v, false, b[t].v,
                                                             (short)0, acc[t], false, false);
    }
    if (K32 < K) {   // zero-padded K tail (true zeros: contributes to accumulator)
        V16 a;
        #pragma unroll
        for (int j = 0; j < 8; ++j) {
            int ka = K32 + 16 * half + j;
            int kb = K32 + 16 * half + 8 + j;
            a.v[j]     = (ka < K) ? Arow[ka] : (bf16_t)0.0f;
            a.v[j + 8] = (kb < K) ? Arow[kb] : (bf16_t)0.0f;
        }
        #pragma unroll
        for (int t = 0; t < NT; ++t) {
            V16 b;
            #pragma unroll
            for (int j = 0; j < 16; ++j) {
                int kk = K32 + 16 * half + j;
                bf16_t bv = (bf16_t)0.0f;
                if (kk < K) bv = TRANSB ? Bcol[t][kk] : Bp[(long)kk * ldb + colB[t]];
                b.v[j] = bv;
            }
            acc[t] = __builtin_amdgcn_wmma_f32_16x16x32_bf16(false, a.v, false, b.v,
                                                             (short)0, acc[t], false, false);
        }
    }

    #pragma unroll
    for (int t = 0; t < NT; ++t) {
        if (!colOk[t]) continue;
        const int col = n0 + t * 16 + l16;
        #pragma unroll
        for (int r = 0; r < 8; ++r) {
            int orow = m0 + r + 8 * half;      // C/D layout: vgpr r -> M = r + 8*half
            if (orow >= M) continue;
            float v = acc[t][r];
            if (flags & 1) v += bias[col];
            if (flags & 2) v = 0.5f * v * (1.0f + erff(v * 0.70710678118654752f));
            long o = dOff + (long)orow * ldd + col;
            if (flags & 4) v += resid[o];
            if (Df) Df[o] = v;
            if (Db) Db[o] = (bf16_t)v;
        }
    }
}

// ---------------------------------------------------------------------------
// fp32 -> bf16 convert + transpose:  Wt[m][n*K+k] = W[m][k*N+n]
// ---------------------------------------------------------------------------
__global__ void convert_transpose_kernel(const float* W, bf16_t* Wt, int K, int N, long nmat)
{
    long i = (long)blockIdx.x * blockDim.x + threadIdx.x;
    long kn = (long)K * N;
    if (i >= nmat * kn) return;
    long m = i / kn;
    long r = i - m * kn;
    int  n = (int)(r % N);
    int  k = (int)(r / N);
    Wt[m * kn + (long)n * K + k] = (bf16_t)W[m * kn + (long)k * N + n];
}

// ---------------------------------------------------------------------------
// patchify: x[B,3,224,224] -> P[B*196, 768] bf16, feature = c*256 + py*16 + px
// ---------------------------------------------------------------------------
__global__ void patchify_kernel(const float* x, bf16_t* P)
{
    long i = (long)blockIdx.x * blockDim.x + threadIdx.x;
    long total = (long)NB * NPATCH_ * PDIM;
    if (i >= total) return;
    int f = (int)(i % PDIM);
    long t = i / PDIM;
    int p = (int)(t % NPATCH_);
    int b = (int)(t / NPATCH_);
    int c = f >> 8, rem = f & 255, py = rem >> 4, px = rem & 15;
    int gy = p / 14, gx = p % 14;
    long src = (((long)(b * 3 + c) * 224 + gy * 16 + py) * 224) + gx * 16 + px;
    P[i] = (bf16_t)x[src];
}

// X[B,197,C] = concat(cls, patch_embed) + pos
__global__ void assemble_kernel(const float* PE, const float* cls, const float* pos, float* X)
{
    long i = (long)blockIdx.x * blockDim.x + threadIdx.x;
    long total = (long)NB * NP0 * CDIM;
    if (i >= total) return;
    int c = (int)(i % CDIM);
    long t = i / CDIM;
    int n = (int)(t % NP0);
    int b = (int)(t / NP0);
    float v = (n == 0) ? cls[c] : PE[((long)b * NPATCH_ + (n - 1)) * CDIM + c];
    X[i] = v + pos[(long)n * CDIM + c];
}

// one wave per row LayerNorm, fp32 in -> bf16 out
__global__ __launch_bounds__(256)
void layernorm_kernel(const float* X, const float* g, const float* bta,
                      bf16_t* Yb, int rows, long inStride, long outStride)
{
    int wid  = (blockIdx.x * blockDim.x + threadIdx.x) >> 5;
    int lane = threadIdx.x & 31;
    if (wid >= rows) return;
    const float* x = X + (long)wid * inStride;
    float s = 0.f, s2 = 0.f;
    for (int c = lane; c < CDIM; c += 32) { float v = x[c]; s += v; s2 += v * v; }
    for (int o = 16; o > 0; o >>= 1) { s += __shfl_xor(s, o, 32); s2 += __shfl_xor(s2, o, 32); }
    float m = s * (1.0f / CDIM);
    float var = s2 * (1.0f / CDIM) - m * m;
    float r = rsqrtf(var + 1e-6f);
    for (int c = lane; c < CDIM; c += 32)
        Yb[(long)wid * outStride + c] = (bf16_t)((x[c] - m) * r * g[c] + bta[c]);
}

// one wave per row softmax(scale*s), fp32 logits -> bf16 probs (padded stride)
__global__ __launch_bounds__(256)
void softmax_kernel(const float* S, bf16_t* P, int rows, int n, int ldS, int ldP, float scale)
{
    int wid  = (blockIdx.x * blockDim.x + threadIdx.x) >> 5;
    int lane = threadIdx.x & 31;
    if (wid >= rows) return;
    const float* s = S + (long)wid * ldS;
    float mx = -3.0e38f;
    for (int c = lane; c < n; c += 32) mx = fmaxf(mx, s[c]);
    for (int o = 16; o > 0; o >>= 1) mx = fmaxf(mx, __shfl_xor(mx, o, 32));
    float sum = 0.f;
    for (int c = lane; c < n; c += 32) sum += __expf((s[c] - mx) * scale);
    for (int o = 16; o > 0; o >>= 1) sum += __shfl_xor(sum, o, 32);
    float inv = 1.0f / sum;
    for (int c = lane; c < n; c += 32)
        P[(long)wid * ldP + c] = (bf16_t)(__expf((s[c] - mx) * scale) * inv);
}

// scores[b,j] = mean_h attn[b,h,0,1+j]
__global__ void scores_kernel(const bf16_t* P, float* scores, int N, int ldP, int nTok)
{
    int i = blockIdx.x * blockDim.x + threadIdx.x;
    if (i >= NB * nTok) return;
    int j = i % nTok, b = i / nTok;
    float s = 0.f;
    for (int h = 0; h < NH; ++h)
        s += (float)P[(long)((b * NH + h) * N) * ldP + 1 + j];
    scores[b * NPATCH_ + j] = s * (1.0f / NH);
}

// one wave per batch: iterative argmax top-k (desc value, asc index on ties)
__global__ __launch_bounds__(32)
void topk_kernel(const float* scores, int* idx, int nTok, int keep)
{
    int b = blockIdx.x, lane = threadIdx.x;
    const float* s = scores + (long)b * NPATCH_;
    float loc[7];
    #pragma unroll
    for (int j = 0; j < 7; ++j) {
        int t = j * 32 + lane;
        loc[j] = (t < nTok) ? s[t] : -3.0e38f;
    }
    for (int it = 0; it < keep; ++it) {
        float bv = -3.0e38f; int bi = 1 << 20;
        #pragma unroll
        for (int j = 0; j < 7; ++j) {
            int t = j * 32 + lane;
            if (loc[j] > bv || (loc[j] == bv && t < bi)) { bv = loc[j]; bi = t; }
        }
        for (int o = 16; o > 0; o >>= 1) {
            float ov = __shfl_xor(bv, o, 32);
            int   oi = __shfl_xor(bi, o, 32);
            if (ov > bv || (ov == bv && oi < bi)) { bv = ov; bi = oi; }
        }
        if (lane == 0) idx[b * NPATCH_ + it] = bi;
        if ((bi & 31) == lane) {
            int sj = bi >> 5;
            #pragma unroll
            for (int j = 0; j < 7; ++j) if (j == sj) loc[j] = -3.0e38f;
        }
    }
}

// Y[b,0]=X[b,0]; Y[b,1+i]=X[b,1+idx[b][i]]
__global__ void gather_kernel(const float* X, const int* idx, float* Y, int Nold, int keep)
{
    long i = (long)blockIdx.x * blockDim.x + threadIdx.x;
    int Nnew = keep + 1;
    long total = (long)NB * Nnew * CDIM;
    if (i >= total) return;
    int c = (int)(i % CDIM);
    long t = i / CDIM;
    int n = (int)(t % Nnew);
    int b = (int)(t / Nnew);
    long src;
    if (n == 0) src = (long)b * Nold * CDIM + c;
    else {
        int tok = idx[b * NPATCH_ + (n - 1)];
        src = ((long)b * Nold + 1 + tok) * CDIM + c;
    }
    Y[i] = X[src];
}

// ---------------------------------------------------------------------------
static void launch_gemm(hipStream_t st, bool transB, int nt,
                        const bf16_t* A, const bf16_t* Bp,
                        const float* bias, const float* resid, float* Df, bf16_t* Db,
                        int M, int N, int K, int lda, int ldb, int ldd, int flags,
                        int batches, int zdiv,
                        long sA1, long sA2, long sB1, long sB2, long sD1, long sD2)
{
    dim3 grid((N + 16 * nt - 1) / (16 * nt), (M + 15) / 16, batches);
    if (transB) {
        if (nt == 4)
            wmma_gemm_bf16<1, 4><<<grid, 32, 0, st>>>(A, Bp, bias, resid, Df, Db, M, N, K,
                lda, ldb, ldd, flags, zdiv, sA1, sA2, sB1, sB2, sD1, sD2);
        else
            wmma_gemm_bf16<1, 2><<<grid, 32, 0, st>>>(A, Bp, bias, resid, Df, Db, M, N, K,
                lda, ldb, ldd, flags, zdiv, sA1, sA2, sB1, sB2, sD1, sD2);
    } else {
        if (nt == 4)
            wmma_gemm_bf16<0, 4><<<grid, 32, 0, st>>>(A, Bp, bias, resid, Df, Db, M, N, K,
                lda, ldb, ldd, flags, zdiv, sA1, sA2, sB1, sB2, sD1, sD2);
        else
            wmma_gemm_bf16<0, 2><<<grid, 32, 0, st>>>(A, Bp, bias, resid, Df, Db, M, N, K,
                lda, ldb, ldd, flags, zdiv, sA1, sA2, sB1, sB2, sD1, sD2);
    }
}

extern "C" void kernel_launch(void* const* d_in, const int* in_sizes, int n_in,
                              void* d_out, int out_size, void* d_ws, size_t ws_size,
                              hipStream_t stream)
{
    (void)in_sizes; (void)n_in; (void)out_size; (void)ws_size;
    const float* x       = (const float*)d_in[0];
    const float* patch_w = (const float*)d_in[1];
    const float* patch_b = (const float*)d_in[2];
    const float* cls_tok = (const float*)d_in[3];
    const float* pos     = (const float*)d_in[4];
    const float* ln1_g   = (const float*)d_in[5];
    const float* ln1_b   = (const float*)d_in[6];
    const float* qkv_w   = (const float*)d_in[7];
    const float* qkv_b   = (const float*)d_in[8];
    const float* proj_w  = (const float*)d_in[9];
    const float* proj_b  = (const float*)d_in[10];
    const float* ln2_g   = (const float*)d_in[11];
    const float* ln2_b   = (const float*)d_in[12];
    const float* fc1_w   = (const float*)d_in[13];
    const float* fc1_b   = (const float*)d_in[14];
    const float* fc2_w   = (const float*)d_in[15];
    const float* fc2_b   = (const float*)d_in[16];
    const float* norm_g  = (const float*)d_in[17];
    const float* norm_b  = (const float*)d_in[18];
    const float* head_w  = (const float*)d_in[19];
    const float* head_b  = (const float*)d_in[20];

    char* ws = (char*)d_ws; size_t off = 0;
    auto alloc = [&](size_t bytes) -> void* {
        off = (off + 255) & ~(size_t)255;
        void* p = ws + off; off += bytes; return p;
    };
    // bf16 transposed weight caches
    bf16_t* WQKV  = (bf16_t*)alloc((size_t)NLAYER * CDIM * 3 * CDIM * 2);
    bf16_t* WPROJ = (bf16_t*)alloc((size_t)NLAYER * CDIM * CDIM * 2);
    bf16_t* WFC1  = (bf16_t*)alloc((size_t)NLAYER * CDIM * MLPD * 2);
    bf16_t* WFC2  = (bf16_t*)alloc((size_t)NLAYER * MLPD * CDIM * 2);
    bf16_t* WPAT  = (bf16_t*)alloc((size_t)PDIM * CDIM * 2);
    bf16_t* WHEAD = (bf16_t*)alloc((size_t)CDIM * NCLS_ * 2);
    // activations
    float*  X0     = (float*) alloc((size_t)NB * NP0 * CDIM * 4);
    float*  X1     = (float*) alloc((size_t)NB * NP0 * CDIM * 4);   // also patch-embed temp
    bf16_t* YBF    = (bf16_t*)alloc((size_t)NB * NP0 * CDIM * 2);
    bf16_t* QKVBF  = (bf16_t*)alloc((size_t)NB * NP0 * 3 * CDIM * 2);
    float*  ATTN   = (float*) alloc((size_t)NB * NH * NP0 * NP0 * 4);
    bf16_t* ATTNBF = (bf16_t*)alloc((size_t)NB * NH * NP0 * 200 * 2);
    bf16_t* AOUTBF = (bf16_t*)alloc((size_t)NB * NP0 * CDIM * 2);
    bf16_t* MLPHBF = (bf16_t*)alloc((size_t)NB * NP0 * MLPD * 2);   // also patchify temp
    float*  SCORES = (float*) alloc((size_t)NB * NPATCH_ * 4);
    int*    IDX    = (int*)   alloc((size_t)NB * NPATCH_ * 4);
    bf16_t* CLSBF  = (bf16_t*)alloc((size_t)NB * CDIM * 2);

    // ---- weight conversion (fp32 -> bf16, transposed) ----
    auto conv = [&](const float* W, bf16_t* Wt, int K, int Ncols, long nmat) {
        long tot = nmat * (long)K * Ncols;
        convert_transpose_kernel<<<(unsigned)((tot + 255) / 256), 256, 0, stream>>>(W, Wt, K, Ncols, nmat);
    };
    conv(qkv_w,  WQKV,  CDIM, 3 * CDIM, NLAYER);
    conv(proj_w, WPROJ, CDIM, CDIM,     NLAYER);
    conv(fc1_w,  WFC1,  CDIM, MLPD,     NLAYER);
    conv(fc2_w,  WFC2,  MLPD, CDIM,     NLAYER);
    conv(patch_w, WPAT, PDIM, CDIM, 1);
    conv(head_w, WHEAD, CDIM, NCLS_, 1);

    // ---- patch embed ----
    bf16_t* PATCHBF = MLPHBF;                     // reuse (disjoint in time)
    float*  PEOUT   = X1;                         // reuse
    long tp = (long)NB * NPATCH_ * PDIM;
    patchify_kernel<<<(unsigned)((tp + 255) / 256), 256, 0, stream>>>(x, PATCHBF);
    launch_gemm(stream, true, 4, PATCHBF, WPAT, patch_b, nullptr, PEOUT, nullptr,
                NB * NPATCH_, CDIM, PDIM, PDIM, PDIM, CDIM, 1, 1, 1, 0, 0, 0, 0, 0, 0);
    long ta = (long)NB * NP0 * CDIM;
    assemble_kernel<<<(unsigned)((ta + 255) / 256), 256, 0, stream>>>(PEOUT, cls_tok, pos, X0);

    // ---- transformer layers ----
    float* Xcur = X0; float* Xalt = X1;
    int N = NP0;
    for (int l = 0; l < NLAYER; ++l) {
        int M = NB * N;
        int Npad = (N + 7) & ~7;
        // LN1
        layernorm_kernel<<<(M + 7) / 8, 256, 0, stream>>>(Xcur, ln1_g + l * CDIM, ln1_b + l * CDIM,
                                                          YBF, M, CDIM, CDIM);
        // QKV = LN1(x) @ qkv_w + b   -> bf16
        launch_gemm(stream, true, 4, YBF, WQKV + (size_t)l * CDIM * 3 * CDIM, qkv_b + l * 3 * CDIM,
                    nullptr, nullptr, QKVBF, M, 3 * CDIM, CDIM, CDIM, CDIM, 3 * CDIM,
                    1, 1, 1, 0, 0, 0, 0, 0, 0);
        // logits = Q @ K^T  (batched over b,h) -> fp32
        launch_gemm(stream, true, 2, QKVBF, QKVBF + CDIM, nullptr, nullptr, ATTN, nullptr,
                    N, N, HDIM, 3 * CDIM, 3 * CDIM, N, 0, NB * NH, NH,
                    (long)N * 3 * CDIM, HDIM, (long)N * 3 * CDIM, HDIM,
                    (long)NH * N * N, (long)N * N);
        // softmax (scale folded) -> bf16 probs, padded stride
        int rows = NB * NH * N;
        softmax_kernel<<<(rows + 7) / 8, 256, 0, stream>>>(ATTN, ATTNBF, rows, N, N, Npad, 0.125f);
        // out = P @ V -> bf16
        launch_gemm(stream, false, 4, ATTNBF, QKVBF + 2 * CDIM, nullptr, nullptr, nullptr, AOUTBF,
                    N, HDIM, N, Npad, 3 * CDIM, CDIM, 0, NB * NH, NH,
                    (long)NH * N * Npad, (long)N * Npad, (long)N * 3 * CDIM, HDIM,
                    (long)N * CDIM, HDIM);
        // x += out @ proj_w + b   (fp32 residual stream)
        launch_gemm(stream, true, 4, AOUTBF, WPROJ + (size_t)l * CDIM * CDIM, proj_b + l * CDIM,
                    Xcur, Xcur, nullptr, M, CDIM, CDIM, CDIM, CDIM, CDIM,
                    1 | 4, 1, 1, 0, 0, 0, 0, 0, 0);
        // token prune
        if (l == 2 || l == 4 || l == 6) {
            int nTok = N - 1;
            int keep = (l == 2) ? 176 : (l == 4) ? 149 : 119;  // int(nTok*(1-ratio))
            int tot = NB * nTok;
            scores_kernel<<<(tot + 255) / 256, 256, 0, stream>>>(ATTNBF, SCORES, N, Npad, nTok);
            topk_kernel<<<NB, 32, 0, stream>>>(SCORES, IDX, nTok, keep);
            long tg = (long)NB * (keep + 1) * CDIM;
            gather_kernel<<<(unsigned)((tg + 255) / 256), 256, 0, stream>>>(Xcur, IDX, Xalt, N, keep);
            float* t = Xcur; Xcur = Xalt; Xalt = t;
            N = keep + 1; M = NB * N;
        }
        // LN2
        layernorm_kernel<<<(M + 7) / 8, 256, 0, stream>>>(Xcur, ln2_g + l * CDIM, ln2_b + l * CDIM,
                                                          YBF, M, CDIM, CDIM);
        // h = gelu(LN2(x) @ fc1 + b) -> bf16
        launch_gemm(stream, true, 4, YBF, WFC1 + (size_t)l * CDIM * MLPD, fc1_b + l * MLPD,
                    nullptr, nullptr, MLPHBF, M, MLPD, CDIM, CDIM, CDIM, MLPD,
                    1 | 2, 1, 1, 0, 0, 0, 0, 0, 0);
        // x += h @ fc2 + b
        launch_gemm(stream, true, 4, MLPHBF, WFC2 + (size_t)l * MLPD * CDIM, fc2_b + l * CDIM,
                    Xcur, Xcur, nullptr, M, CDIM, MLPD, MLPD, MLPD, CDIM,
                    1 | 4, 1, 1, 0, 0, 0, 0, 0, 0);
    }

    // ---- final LN on CLS rows only + classifier head ----
    layernorm_kernel<<<(NB + 7) / 8, 256, 0, stream>>>(Xcur, norm_g, norm_b, CLSBF,
                                                       NB, (long)N * CDIM, CDIM);
    launch_gemm(stream, true, 2, CLSBF, WHEAD, head_b, nullptr, (float*)d_out, nullptr,
                NB, NCLS_, CDIM, CDIM, CDIM, NCLS_, 1, 1, 1, 0, 0, 0, 0, 0, 0);
}
